// MultiQueryAttentionLayerV2_46016279610270
// MI455X (gfx1250) — compile-verified, matrix operand order checked by reading, not run on hardware
//
#include <hip/hip_runtime.h>

// ---------------------------------------------------------------------------
// Types for CDNA5 WMMA (wave32): V_WMMA_F32_16X16X32_BF16
// ---------------------------------------------------------------------------
typedef __attribute__((ext_vector_type(16))) __bf16 v16bf;
typedef __attribute__((ext_vector_type(8)))  __bf16 bf8_t;
typedef __attribute__((ext_vector_type(8)))  float  v8f;
typedef __attribute__((ext_vector_type(4)))  float  f32x4;

// Problem constants
constexpr int B_ = 2, D_ = 512, N_ = 2048, H_ = 8, K_ = 64, V_ = 64;
constexpr int HV = H_ * V_;            // 512

// ---------------------------------------------------------------------------
// Fragment loaders.
// 16-bit A/B WMMA layout (ISA 7.12.2): lane holds row/col = lane%16; its 16
// elements are two contiguous runs of 8 along K, at K-offsets r0 and r0+16
// where r0 = (lane>=16 ? 8 : 0).
// ---------------------------------------------------------------------------
__device__ inline v16bf load_frag_bf16(const __bf16* p, int r0) {
  bf8_t lo = *(const bf8_t*)(p + r0);
  bf8_t hv = *(const bf8_t*)(p + r0 + 16);
  v16bf f;
#pragma unroll
  for (int i = 0; i < 8; ++i) { f[i] = lo[i]; f[i + 8] = hv[i]; }
  return f;
}

__device__ inline v16bf load_frag_f32_contig(const float* p, int r0) {
  f32x4 a0 = *(const f32x4*)(p + r0);
  f32x4 a1 = *(const f32x4*)(p + r0 + 4);
  f32x4 b0 = *(const f32x4*)(p + r0 + 16);
  f32x4 b1 = *(const f32x4*)(p + r0 + 20);
  v16bf f;
#pragma unroll
  for (int i = 0; i < 4; ++i) {
    f[i]      = (__bf16)a0[i];
    f[i + 4]  = (__bf16)a1[i];
    f[i + 8]  = (__bf16)b0[i];
    f[i + 12] = (__bf16)b1[i];
  }
  return f;
}

__device__ inline v16bf load_frag_f32_strided(const float* p, int r0, int stride) {
  v16bf f;
#pragma unroll
  for (int i = 0; i < 8; ++i) {
    f[i]     = (__bf16)p[(size_t)(r0 + i) * stride];
    f[i + 8] = (__bf16)p[(size_t)(r0 + 16 + i) * stride];
  }
  return f;
}

__device__ inline v8f wmma_bf16(v16bf a, v16bf b, v8f c) {
  return __builtin_amdgcn_wmma_f32_16x16x32_bf16(false, a, false, b, (short)0, c,
                                                 false, false);
}

// ---------------------------------------------------------------------------
// Kernel 1a: Q projection, register-blocked 16x64 per wave (branch-free loop).
//   Qbf(b, n, c) = sum_d x[b][d][n] * query_proj[c][d],  c = h*64+k in [0,512)
// The strided X^T A-fragment is loaded once per k-step and feeds 4 WMMAs.
// ---------------------------------------------------------------------------
__global__ __launch_bounds__(32) void k_project_q(
    const float* __restrict__ x, const float* __restrict__ qp,
    __bf16* __restrict__ Qbf) {
  const int nt = blockIdx.x;            // 0..127  (N/16)
  const int cg = blockIdx.y;            // 0..7    (512/64)
  const int b  = blockIdx.z;
  const int lane = threadIdx.x;
  const int col  = lane & 15;
  const int hi   = lane >> 4;
  const int r0   = hi * 8;
  const int n0   = nt * 16;
  const int nrow = n0 + col;            // A-matrix row (query position)
  const int c0   = cg * 64 + col;       // first of 4 output columns

  const float* xb = x + (size_t)b * D_ * N_;
  v8f acc[4] = {};

  for (int d0 = 0; d0 < D_; d0 += 32) {
    const v16bf A = load_frag_f32_strided(xb + (size_t)d0 * N_ + nrow, r0, N_);
#pragma unroll
    for (int t = 0; t < 4; ++t) {
      v16bf Bf = load_frag_f32_contig(qp + (size_t)(c0 + t * 16) * D_ + d0, r0);
      acc[t] = wmma_bf16(A, Bf, acc[t]);
    }
  }

#pragma unroll
  for (int t = 0; t < 4; ++t) {
#pragma unroll
    for (int j = 0; j < 8; ++j) {        // C row = j + 8*hi, C col = c0 + t*16
      const int n = n0 + j + 8 * hi;
      Qbf[((size_t)b * N_ + n) * HV + c0 + t * 16] = (__bf16)acc[t][j];
    }
  }
}

// ---------------------------------------------------------------------------
// Kernel 1b: K/V projections (both are (D,64) row-major weights).
//   y==0: Kbf(b, m, k) = sum_d x[b][d][m] * key_proj[d][k]
//   y==1: Vt (b, v, m) = sum_d x[b][d][m] * value_proj[d][v]   (transposed)
// Weight base pointer is selected once before the loop; the k-loop is
// branch-free. Only the epilogue picks the destination layout.
// ---------------------------------------------------------------------------
__global__ __launch_bounds__(32) void k_project_kv(
    const float* __restrict__ x, const float* __restrict__ kp,
    const float* __restrict__ vp, __bf16* __restrict__ Kbf,
    __bf16* __restrict__ Vt) {
  const int nt  = blockIdx.x;           // 0..127
  const int sel = blockIdx.y;           // 0 = K, 1 = V
  const int b   = blockIdx.z;
  const int lane = threadIdx.x;
  const int col  = lane & 15;
  const int hi   = lane >> 4;
  const int r0   = hi * 8;
  const int n0   = nt * 16;
  const int nrow = n0 + col;

  const float* wp = (sel == 0) ? kp : vp;   // (D, 64) row-major
  const float* xb = x + (size_t)b * D_ * N_;
  v8f acc[4] = {};

  for (int d0 = 0; d0 < D_; d0 += 32) {
    const v16bf A = load_frag_f32_strided(xb + (size_t)d0 * N_ + nrow, r0, N_);
#pragma unroll
    for (int t = 0; t < 4; ++t) {
      v16bf Bf = load_frag_f32_strided(wp + (size_t)d0 * 64 + t * 16 + col, r0, 64);
      acc[t] = wmma_bf16(A, Bf, acc[t]);
    }
  }

  if (sel == 0) {
#pragma unroll
    for (int t = 0; t < 4; ++t)
#pragma unroll
      for (int j = 0; j < 8; ++j) {
        const int n = n0 + j + 8 * hi;
        Kbf[((size_t)b * N_ + n) * K_ + t * 16 + col] = (__bf16)acc[t][j];
      }
  } else {
#pragma unroll
    for (int t = 0; t < 4; ++t)
#pragma unroll
      for (int j = 0; j < 8; ++j) {
        const int n = n0 + j + 8 * hi;
        Vt[((size_t)b * V_ + t * 16 + col) * (size_t)N_ + n] = (__bf16)acc[t][j];
      }
  }
}

// ---------------------------------------------------------------------------
// Kernel 2: flash attention. One wave per (b, h, 16-row query tile).
// Scores S(n,m) via WMMA; online softmax over m in chunks of 128; P goes
// through a private LDS tile to become B-fragments of P^T; the output is
// accumulated transposed: O^T(v,n) = V^T(v,m) x P^T(m,n), so softmax rescale
// is a per-lane scalar (lane = query column).
// ---------------------------------------------------------------------------
__global__ __launch_bounds__(32) void k_attn(
    const __bf16* __restrict__ Qbf, const __bf16* __restrict__ Kbf,
    const __bf16* __restrict__ Vt, __bf16* __restrict__ Obf) {
  __shared__ __bf16 Plds[16 * 128];      // P chunk, row = query, col = m
  __shared__ float  stats[16];           // row->lane transposer

  const int nt = blockIdx.x;             // 0..127
  const int h  = blockIdx.y;             // 0..7
  const int b  = blockIdx.z;
  const int lane = threadIdx.x;
  const int col  = lane & 15;
  const int hi   = lane >> 4;
  const int r0   = hi * 8;
  const int n0   = nt * 16;

  const float csc = 0.125f * 1.44269504088896340736f;  // (1/sqrt(64))*log2(e)

  // Q fragments for this tile: rows n0..n0+15, k = 0..63 (two 16x32 frags)
  const __bf16* qbase = Qbf + ((size_t)b * N_ + (n0 + col)) * HV + h * K_;
  const v16bf A0 = load_frag_bf16(qbase, r0);
  const v16bf A1 = load_frag_bf16(qbase + 32, r0);

  float M[8], L[8];
#pragma unroll
  for (int j = 0; j < 8; ++j) { M[j] = -3.0e38f; L[j] = 0.f; }
  v8f acc[4] = {};                       // O^T: rows v = vt*16 + j + 8*hi, col n

  for (int mc = 0; mc < N_ / 128; ++mc) {
    // ---- scores for this 128-wide chunk (8 tiles x 2 WMMA) ----
    v8f S[8];
#pragma unroll
    for (int t = 0; t < 8; ++t) {
      const int m0 = mc * 128 + t * 16;
      const __bf16* kb = Kbf + ((size_t)b * N_ + (m0 + col)) * K_;
      v16bf B0 = load_frag_bf16(kb, r0);
      v16bf B1 = load_frag_bf16(kb + 32, r0);
      v8f s = {};
      s = wmma_bf16(A0, B0, s);
      s = wmma_bf16(A1, B1, s);
      S[t] = s;
    }
    // ---- per-row chunk max (row lives across one 16-lane half) ----
    float cm[8];
#pragma unroll
    for (int j = 0; j < 8; ++j) {
      float v = S[0][j];
#pragma unroll
      for (int t = 1; t < 8; ++t) v = fmaxf(v, S[t][j]);
#pragma unroll
      for (int msk = 1; msk < 16; msk <<= 1) v = fmaxf(v, __shfl_xor(v, msk, 32));
      cm[j] = v;
    }
    float alpha[8], rs[8];
#pragma unroll
    for (int j = 0; j < 8; ++j) {
      const float nm = fmaxf(M[j], cm[j]);
      alpha[j] = __builtin_exp2f((M[j] - nm) * csc);
      M[j] = nm;
      rs[j] = 0.f;
    }
    // ---- P = exp2((S - M)*csc): row-sums + bf16 into LDS ----
#pragma unroll
    for (int t = 0; t < 8; ++t) {
#pragma unroll
      for (int j = 0; j < 8; ++j) {
        const float p = __builtin_exp2f((S[t][j] - M[j]) * csc);
        rs[j] += p;
        Plds[(j + 8 * hi) * 128 + t * 16 + col] = (__bf16)p;
      }
    }
#pragma unroll
    for (int j = 0; j < 8; ++j) {
#pragma unroll
      for (int msk = 1; msk < 16; msk <<= 1) rs[j] += __shfl_xor(rs[j], msk, 32);
      L[j] = L[j] * alpha[j] + rs[j];
    }
    // ---- broadcast alpha (indexed by row) to lanes (indexed by query col) ----
    if (lane == 0) {
#pragma unroll
      for (int j = 0; j < 8; ++j) stats[j] = alpha[j];
    }
    if (lane == 16) {
#pragma unroll
      for (int j = 0; j < 8; ++j) stats[8 + j] = alpha[j];
    }
    asm volatile("s_wait_dscnt 0" ::: "memory");   // same-wave LDS RAW fence
    const float av = stats[col];
#pragma unroll
    for (int vt = 0; vt < 4; ++vt)
#pragma unroll
      for (int j = 0; j < 8; ++j) acc[vt][j] *= av;

    // ---- O^T += V^T x P^T  (4 k-steps of 32 x 4 v-tiles) ----
#pragma unroll
    for (int kc = 0; kc < 4; ++kc) {
      bf8_t p0 = *(const bf8_t*)(Plds + col * 128 + kc * 32 + r0);
      bf8_t p1 = *(const bf8_t*)(Plds + col * 128 + kc * 32 + r0 + 16);
      v16bf BP;
#pragma unroll
      for (int i = 0; i < 8; ++i) { BP[i] = p0[i]; BP[i + 8] = p1[i]; }
#pragma unroll
      for (int vt = 0; vt < 4; ++vt) {
        const __bf16* vb =
            Vt + ((size_t)b * V_ + vt * 16 + col) * (size_t)N_ + mc * 128 + kc * 32;
        v16bf AV = load_frag_bf16(vb, r0);
        acc[vt] = wmma_bf16(AV, BP, acc[vt]);
      }
    }
  }

  // ---- normalize by row-sum (transpose L row->lane via LDS) and store O ----
  if (lane == 0) {
#pragma unroll
    for (int j = 0; j < 8; ++j) stats[j] = L[j];
  }
  if (lane == 16) {
#pragma unroll
    for (int j = 0; j < 8; ++j) stats[8 + j] = L[j];
  }
  asm volatile("s_wait_dscnt 0" ::: "memory");
  const float linv = 1.0f / stats[col];

#pragma unroll
  for (int vt = 0; vt < 4; ++vt) {
    bf8_t ov;
#pragma unroll
    for (int j = 0; j < 8; ++j) ov[j] = (__bf16)(acc[vt][j] * linv);
    // O layout (b, n, h*64 + v); elements j cover v = vt*16 + hi*8 + j
    *(bf8_t*)(Obf + ((size_t)b * N_ + n0 + col) * HV + h * V_ + vt * 16 + hi * 8) = ov;
  }
}

// ---------------------------------------------------------------------------
// Kernel 3: output projection, register-blocked 16x64 per wave.
//   out(b,d,n) = sum_{hv} W(d,hv) * O(b,n,hv),  512-deep bf16 WMMA GEMM.
// The fp32 W A-fragment (loads + cvts) is loaded once per k-step and reused
// across 4 n-tiles with cheap contiguous bf16 B-fragments.
// ---------------------------------------------------------------------------
__global__ __launch_bounds__(32) void k_output(
    const __bf16* __restrict__ Obf, const float* __restrict__ Wp,
    float* __restrict__ out) {
  const int ng = blockIdx.x;             // 0..31  (N/64)
  const int dt = blockIdx.y;             // 0..31  (D/16)
  const int b  = blockIdx.z;
  const int lane = threadIdx.x;
  const int col  = lane & 15;
  const int hi   = lane >> 4;
  const int r0   = hi * 8;
  const int n0   = ng * 64;
  const int d0   = dt * 16;

  const float* wb = Wp + (size_t)(d0 + col) * HV;                // A: row = d
  const __bf16* ob0 = Obf + ((size_t)b * N_ + (n0 + col)) * HV;  // B: col = n

  v8f acc[4] = {};
  for (int k0 = 0; k0 < HV; k0 += 32) {
    const v16bf A = load_frag_f32_contig(wb + k0, r0);
#pragma unroll
    for (int t = 0; t < 4; ++t) {
      v16bf Bf = load_frag_bf16(ob0 + (size_t)t * 16 * HV + k0, r0);
      acc[t] = wmma_bf16(A, Bf, acc[t]);
    }
  }
#pragma unroll
  for (int t = 0; t < 4; ++t) {
#pragma unroll
    for (int j = 0; j < 8; ++j) {
      const int d = d0 + j + 8 * hi;
      out[(size_t)b * D_ * N_ + (size_t)d * N_ + n0 + t * 16 + col] = acc[t][j];
    }
  }
}

// ---------------------------------------------------------------------------
extern "C" void kernel_launch(void* const* d_in, const int* in_sizes, int n_in,
                              void* d_out, int out_size, void* d_ws, size_t ws_size,
                              hipStream_t stream) {
  (void)in_sizes; (void)n_in; (void)out_size; (void)ws_size;
  const float* x  = (const float*)d_in[0];
  const float* qp = (const float*)d_in[1];
  const float* kp = (const float*)d_in[2];
  const float* vp = (const float*)d_in[3];
  const float* op = (const float*)d_in[4];
  float* out = (float*)d_out;

  char* ws = (char*)d_ws;
  __bf16* Qbf = (__bf16*)(ws);                               // 4 MiB  (B,N,HV)
  __bf16* Kbf = (__bf16*)(ws + (4u << 20));                  // 512 KiB (B,N,K)
  __bf16* Vt  = (__bf16*)(ws + (4u << 20) + (512u << 10));   // 512 KiB (B,V,N)
  __bf16* Obf = (__bf16*)(ws + (5u << 20));                  // 4 MiB  (B,N,HV)

  k_project_q<<<dim3(N_ / 16, HV / 64, B_), dim3(32), 0, stream>>>(x, qp, Qbf);
  k_project_kv<<<dim3(N_ / 16, 2, B_), dim3(32), 0, stream>>>(x, kp, vp, Kbf, Vt);
  k_attn<<<dim3(N_ / 16, H_, B_), dim3(32), 0, stream>>>(Qbf, Kbf, Vt, Obf);
  k_output<<<dim3(N_ / 64, D_ / 16, B_), dim3(32), 0, stream>>>(Obf, op, out);
}
